// MAL_4415226380652
// MI455X (gfx1250) — matrix operand order repaired and checked by reading, not compiled, not run on hardware
//
#include <hip/hip_runtime.h>

// ---------------- problem constants ----------------
#define B_   16
#define NV_  4
#define C_   768
#define N_   784      // 28*28 tokens
#define QK_  96
#define CD_  3072
#define FD_  3136

// ---------------- GEMM tiling ----------------
#define BM 128
#define BN 128
#define BK 32
#define BKP 40        // padded K extent in LDS (bf16 elems); rows stay 16B aligned

typedef __bf16 bf16_t;
typedef bf16_t v16bf __attribute__((ext_vector_type(16)));
typedef float  v8f   __attribute__((ext_vector_type(8)));
typedef int    v4i   __attribute__((ext_vector_type(4)));

union FragBF { v16bf v; unsigned u[8]; };
union FragF  { v8f   v; float    f[8]; };
union Pack16 { uint4 v; unsigned short s[8]; };

__device__ __forceinline__ unsigned short f32_to_bf16(float f) {
    unsigned u = __builtin_bit_cast(unsigned, f);
    u += 0x7fffu + ((u >> 16) & 1u);          // round to nearest even
    return (unsigned short)(u >> 16);
}

// ---- CDNA5 async global->LDS copy (ASYNCcnt path), with sync fallback ----
#if defined(__has_builtin)
#  if __has_builtin(__builtin_amdgcn_global_load_async_to_lds_b128) && \
      __has_builtin(__builtin_amdgcn_s_wait_asynccnt)
#    define HAS_ASYNC_LDS 1
#  endif
#endif
#ifndef HAS_ASYNC_LDS
#  define HAS_ASYNC_LDS 0
#endif

typedef __attribute__((address_space(1))) v4i* gbl_v4i_p;   // global (device) int4*
typedef __attribute__((address_space(3))) v4i* lds_v4i_p;   // LDS int4*

__device__ __forceinline__ void copy16_g2l(unsigned short* lds_dst,
                                           const unsigned short* gsrc) {
#if HAS_ASYNC_LDS
    __builtin_amdgcn_global_load_async_to_lds_b128(
        (gbl_v4i_p)gsrc, (lds_v4i_p)lds_dst, 0, 0);
#else
    *(uint4*)lds_dst = *(const uint4*)gsrc;
#endif
}
__device__ __forceinline__ void wait_g2l() {
#if HAS_ASYNC_LDS
    __builtin_amdgcn_s_wait_asynccnt(0);
#endif
}

// ---------------------------------------------------------------------------
// Batched bf16 WMMA GEMM:  C[z] = alpha * A[z] * B[z] (+bias)
//   A logical [M,K]: (m,k) at A[m*lda+k] (TA=0) or A[k*lda+m] (TA=1)
//   B logical [K,N]: (k,n) at B[k*ldb+n] (TB=0) or B[n*ldb+k] (TB=1)
//   C dense [M,N] per z (f32 and/or bf16); per-z offset = (z/zd)*s
// ---------------------------------------------------------------------------
template <int TA, int TB>
__global__ void __launch_bounds__(256)
wmma_gemm_bf16(const unsigned short* __restrict__ Ag,
               const unsigned short* __restrict__ Bg,
               float* __restrict__ Cf, unsigned short* __restrict__ Cb,
               const float* __restrict__ biasg, int bias_mode,  // 0 none,1 row,2 col
               long long sBias, int zdBias,
               int M, int N, int K, int lda, int ldb,
               long long sA, int zdA, long long sB, int zdB,
               float alpha)
{
    __shared__ unsigned short As[BM * BKP];   // [m][k]
    __shared__ unsigned short Bs[BN * BKP];   // transposed: [n][k]

    const int z = blockIdx.z;
    const unsigned short* A = Ag + (long long)(z / zdA) * sA;
    const unsigned short* B = Bg + (long long)(z / zdB) * sB;
    const float* bias = biasg ? biasg + (long long)(z / zdBias) * sBias : nullptr;

    const int tid   = threadIdx.x;
    const int m0b   = blockIdx.y * BM;
    const int n0b   = blockIdx.x * BN;
    const int lane  = tid & 31;
    const int wave  = tid >> 5;
    const int wm0   = (wave & 3) * 32;   // 4 waves down
    const int wn0   = (wave >> 2) * 64;  // 2 waves across
    const int mlane = lane & 15;
    const int half  = lane >> 4;

    FragF acc[2][4];
    #pragma unroll
    for (int i = 0; i < 2; ++i)
        #pragma unroll
        for (int j = 0; j < 4; ++j)
            #pragma unroll
            for (int r = 0; r < 8; ++r) acc[i][j].f[r] = 0.0f;

    const int nkt = (K + BK - 1) / BK;
    for (int kt = 0; kt < nkt; ++kt) {
        const int k0 = kt * BK;

        if (kt + 1 < nkt) {   // speculative prefetch of next K tiles
            const int kn = k0 + BK;
            __builtin_prefetch(TA ? (A + (size_t)kn * lda + m0b)
                                  : (A + (size_t)m0b * lda + kn), 0, 1);
            __builtin_prefetch(TB ? (B + (size_t)n0b * ldb + kn)
                                  : (B + (size_t)kn * ldb + n0b), 0, 1);
        }

        // ---- stage A tile (BM x BK) in 16B chunks: 512 chunks / 256 threads ----
        if (TA == 0) {
            // contiguous along k -> matches As row layout: async b128 eligible
            for (int c = tid; c < (BM * BK) / 8; c += 256) {
                int m  = c >> 2;
                int k8 = (c & 3) * 8;
                int gm = m0b + m, gk = k0 + k8;
                unsigned short* dst = &As[m * BKP + k8];
                if (gm < M && gk + 7 < K) {
                    copy16_g2l(dst, A + (size_t)gm * lda + gk);
                } else {
                    #pragma unroll
                    for (int e = 0; e < 8; ++e)
                        dst[e] = (gm < M && gk + e < K) ? A[(size_t)gm * lda + gk + e] : 0;
                }
            }
        } else {
            // contiguous along m -> b128 global load, 8 strided LDS stores
            for (int c = tid; c < (BM * BK) / 8; c += 256) {
                int k  = c >> 4;
                int m8 = (c & 15) * 8;
                int gk = k0 + k, gm0 = m0b + m8;
                if (gk < K && gm0 + 7 < M) {
                    Pack16 p; p.v = *(const uint4*)(A + (size_t)gk * lda + gm0);
                    #pragma unroll
                    for (int e = 0; e < 8; ++e) As[(m8 + e) * BKP + k] = p.s[e];
                } else {
                    #pragma unroll
                    for (int e = 0; e < 8; ++e)
                        As[(m8 + e) * BKP + k] =
                            (gk < K && gm0 + e < M) ? A[(size_t)gk * lda + gm0 + e] : 0;
                }
            }
        }
        // ---- stage B tile (BK x BN) transposed into Bs[n][k] ----
        if (TB == 1) {
            // B stored [N,K]: contiguous along k -> async b128 eligible
            for (int c = tid; c < (BN * BK) / 8; c += 256) {
                int n  = c >> 2;
                int k8 = (c & 3) * 8;
                int gn = n0b + n, gk = k0 + k8;
                unsigned short* dst = &Bs[n * BKP + k8];
                if (gn < N && gk + 7 < K) {
                    copy16_g2l(dst, B + (size_t)gn * ldb + gk);
                } else {
                    #pragma unroll
                    for (int e = 0; e < 8; ++e)
                        dst[e] = (gn < N && gk + e < K) ? B[(size_t)gn * ldb + gk + e] : 0;
                }
            }
        } else {
            // B stored [K,N]: contiguous along n -> b128 load, strided LDS stores
            for (int c = tid; c < (BN * BK) / 8; c += 256) {
                int k  = c >> 4;
                int n8 = (c & 15) * 8;
                int gk = k0 + k, gn0 = n0b + n8;
                if (gk < K && gn0 + 7 < N) {
                    Pack16 p; p.v = *(const uint4*)(B + (size_t)gk * ldb + gn0);
                    #pragma unroll
                    for (int e = 0; e < 8; ++e) Bs[(n8 + e) * BKP + k] = p.s[e];
                } else {
                    #pragma unroll
                    for (int e = 0; e < 8; ++e)
                        Bs[(n8 + e) * BKP + k] =
                            (gk < K && gn0 + e < N) ? B[(size_t)gk * ldb + gn0 + e] : 0;
                }
            }
        }
        wait_g2l();          // drain this wave's ASYNCcnt before the barrier
        __syncthreads();

        // ---- fragments: contiguous k = 16*half + s, same order in A and B ----
        FragBF afrag[2], bfrag[4];
        #pragma unroll
        for (int mt = 0; mt < 2; ++mt) {
            const unsigned short* ap = &As[(wm0 + mt * 16 + mlane) * BKP];
            #pragma unroll
            for (int r = 0; r < 8; ++r) {
                int kk = 16 * half + 2 * r;
                afrag[mt].u[r] = *(const unsigned*)(ap + kk);
            }
        }
        #pragma unroll
        for (int nt = 0; nt < 4; ++nt) {
            const unsigned short* bp = &Bs[(wn0 + nt * 16 + mlane) * BKP + 16 * half];
            #pragma unroll
            for (int r = 0; r < 8; ++r)
                bfrag[nt].u[r] = *(const unsigned*)(bp + 2 * r);
        }

        #pragma unroll
        for (int mt = 0; mt < 2; ++mt)
            #pragma unroll
            for (int nt = 0; nt < 4; ++nt)
                acc[mt][nt].v = __builtin_amdgcn_wmma_f32_16x16x32_bf16(
                    false, afrag[mt].v, false, bfrag[nt].v,
                    (short)0, acc[mt][nt].v, false, false);
        __syncthreads();
    }

    // epilogue (C/D layout: VGPR r -> row r + 8*half, col = lane%16)
    const size_t zo = (size_t)z * (size_t)M * (size_t)N;
    #pragma unroll
    for (int mt = 0; mt < 2; ++mt)
        #pragma unroll
        for (int nt = 0; nt < 4; ++nt) {
            int col = n0b + wn0 + nt * 16 + mlane;
            #pragma unroll
            for (int r = 0; r < 8; ++r) {
                int row = m0b + wm0 + mt * 16 + r + 8 * half;
                if (row < M && col < N) {
                    float v = acc[mt][nt].f[r] * alpha;
                    if (bias_mode == 1) v += bias[row];
                    else if (bias_mode == 2) v += bias[col];
                    size_t o = zo + (size_t)row * N + col;
                    if (Cf) Cf[o] = v;
                    if (Cb) Cb[o] = f32_to_bf16(v);
                }
            }
        }
}

// ---------------- elementwise / softmax helpers ----------------
__global__ void cast_f32_bf16_k(const float* __restrict__ in,
                                unsigned short* __restrict__ out, size_t n) {
    size_t i = (size_t)blockIdx.x * 256 + threadIdx.x;
    if (i < n) out[i] = f32_to_bf16(in[i]);
}

__global__ void __launch_bounds__(256)
softmax_rows_k(const float* __restrict__ S, unsigned short* __restrict__ P, int L) {
    __shared__ float red[256];
    const size_t row = blockIdx.x;
    const float* s = S + row * (size_t)L;
    unsigned short* p = P + row * (size_t)L;
    const int tid = threadIdx.x;
    float mx = -3.0e38f;
    for (int i = tid; i < L; i += 256) mx = fmaxf(mx, s[i]);
    red[tid] = mx; __syncthreads();
    for (int off = 128; off > 0; off >>= 1) {
        if (tid < off) red[tid] = fmaxf(red[tid], red[tid + off]);
        __syncthreads();
    }
    mx = red[0]; __syncthreads();
    float sum = 0.0f;
    for (int i = tid; i < L; i += 256) sum += __expf(s[i] - mx);
    red[tid] = sum; __syncthreads();
    for (int off = 128; off > 0; off >>= 1) {
        if (tid < off) red[tid] += red[tid + off];
        __syncthreads();
    }
    const float inv = 1.0f / red[0];
    for (int i = tid; i < L; i += 256) p[i] = f32_to_bf16(__expf(s[i] - mx) * inv);
}

// Z := gamma[v]*Z + X  (self-attention residual; [NV][B][C][N] layout)
__global__ void sa_epilogue_k(float* __restrict__ Z, const float* __restrict__ X,
                              const float* __restrict__ gamma, size_t perView, size_t total) {
    size_t i = (size_t)blockIdx.x * 256 + threadIdx.x;
    if (i >= total) return;
    int v = (int)(i / perView);
    Z[i] = gamma[v] * Z[i] + X[i];
}

// SAout [NV][B][C][N] -> Tca [B][N][NV*C] and Tfa [B][C][NV*N], f32 + bf16
__global__ void repack_sa_k(const float* __restrict__ SA,
                            float* __restrict__ Tca_f, unsigned short* __restrict__ Tca_b,
                            float* __restrict__ Tfa_f, unsigned short* __restrict__ Tfa_b) {
    size_t idx = (size_t)blockIdx.x * 256 + threadIdx.x;
    const size_t total = (size_t)NV_ * B_ * C_ * N_;
    if (idx >= total) return;
    int n = (int)(idx % N_); size_t t = idx / N_;
    int c = (int)(t % C_); t /= C_;
    int b = (int)(t % B_); int v = (int)(t / B_);
    float val = SA[idx];
    size_t ica = ((size_t)b * N_ + n) * CD_ + (size_t)v * C_ + c;
    size_t ifa = ((size_t)b * C_ + c) * FD_ + (size_t)v * N_ + n;
    Tca_f[ica] = val; Tca_b[ica] = f32_to_bf16(val);
    Tfa_f[ifa] = val; Tfa_b[ifa] = f32_to_bf16(val);
}

// bug-faithful  y = swapaxes(Z,1,2).reshape(B,R,D) + X  (in-place on X)
__global__ void scramble_residual_k(const float* __restrict__ Z, float* __restrict__ O,
                                    int R, int D, size_t total) {
    size_t idx = (size_t)blockIdx.x * 256 + threadIdx.x;
    if (idx >= total) return;
    int d = (int)(idx % D); size_t t = idx / D;
    int r = (int)(t % R);  int b = (int)(t / R);
    size_t i = (size_t)r * D + d;
    size_t zi = (size_t)b * R * D + (i % R) * (size_t)D + (i / R);
    O[idx] += Z[zi];
}

// out[b,j,n] = 0.125 * sum_v ( Ofa[b, m/4, (m%4)*N + n] + Oca[b, n, m] ), m = v*C + j
__global__ void final_combine_k(const float* __restrict__ Ofa,
                                const float* __restrict__ Oca,
                                float* __restrict__ out) {
    size_t idx = (size_t)blockIdx.x * 256 + threadIdx.x;
    const size_t total = (size_t)B_ * C_ * N_;
    if (idx >= total) return;
    int n = (int)(idx % N_); size_t t = idx / N_;
    int j = (int)(t % C_);  int b = (int)(t / C_);
    float s = 0.0f;
    #pragma unroll
    for (int v = 0; v < NV_; ++v) {
        int m = v * C_ + j;
        s += Ofa[((size_t)b * C_ + (m >> 2)) * FD_ + (size_t)(m & 3) * N_ + n];
        s += Oca[((size_t)b * N_ + n) * CD_ + m];
    }
    out[idx] = 0.125f * s;
}

// ---------------------------------------------------------------------------
extern "C" void kernel_launch(void* const* d_in, const int* in_sizes, int n_in,
                              void* d_out, int out_size, void* d_ws, size_t ws_size,
                              hipStream_t stream) {
    (void)in_sizes; (void)n_in; (void)out_size; (void)ws_size;
    typedef unsigned short ush;

    const float* features = (const float*)d_in[0];
    const float* sa_wq = (const float*)d_in[1];  const float* sa_bq = (const float*)d_in[2];
    const float* sa_wk = (const float*)d_in[3];  const float* sa_bk = (const float*)d_in[4];
    const float* sa_wv = (const float*)d_in[5];  const float* sa_bv = (const float*)d_in[6];
    const float* sa_gamma = (const float*)d_in[7];
    const float* ca_wq = (const float*)d_in[8];  const float* ca_bq = (const float*)d_in[9];
    const float* ca_wk = (const float*)d_in[10]; const float* ca_bk = (const float*)d_in[11];
    const float* ca_wv = (const float*)d_in[12]; const float* ca_bv = (const float*)d_in[13];
    const float* fa_wq = (const float*)d_in[14]; const float* fa_bq = (const float*)d_in[15];
    const float* fa_wk = (const float*)d_in[16]; const float* fa_bk = (const float*)d_in[17];
    const float* fa_wv = (const float*)d_in[18]; const float* fa_bv = (const float*)d_in[19];
    float* out = (float*)d_out;

    unsigned char* wsb = (unsigned char*)d_ws;
    size_t off = 0;
    auto alloc = [&](size_t bytes) -> void* {
        void* p = wsb + off;
        off = (off + bytes + 255) & ~(size_t)255;
        return p;
    };
    const size_t XN  = (size_t)NV_ * B_ * C_ * N_;
    const size_t EN  = (size_t)NV_ * B_ * N_ * N_;
    const size_t QTN = (size_t)NV_ * B_ * N_ * QK_;
    const size_t TN  = (size_t)B_ * C_ * FD_;
    const size_t SN  = (size_t)B_ * N_ * N_;

    ush*   Xbf   = (ush*)alloc(XN * 2);              // reused as Qfa, then Qca
    ush*   Wsaq  = (ush*)alloc((size_t)NV_*QK_*C_ * 2);
    ush*   Wsak  = (ush*)alloc((size_t)NV_*QK_*C_ * 2);
    ush*   Wsav  = (ush*)alloc((size_t)NV_*C_*C_  * 2);
    ush*   Wcaq  = (ush*)alloc((size_t)CD_*CD_ * 2);
    ush*   Wcak  = (ush*)alloc((size_t)CD_*CD_ * 2);
    ush*   Wcav  = (ush*)alloc((size_t)CD_*CD_ * 2);
    ush*   Wfaq  = (ush*)alloc((size_t)FD_*FD_ * 2);
    ush*   Wfak  = (ush*)alloc((size_t)FD_*FD_ * 2);
    ush*   Wfav  = (ush*)alloc((size_t)FD_*FD_ * 2);
    ush*   Qt    = (ush*)alloc(QTN * 2);
    ush*   Kt    = (ush*)alloc(QTN * 2);
    ush*   AttnS = (ush*)alloc(EN * 2);              // sa attn; reused as Vfa, then Vca
    ush*   Vbf   = (ush*)alloc(XN * 2);              // reused as Kfa, then Kca
    float* Epool = (float*)alloc(EN * 4);            // E_sa / Zsa+SAout / Zfa / Zca
    float* Tca_f = (float*)alloc(TN * 4);            // becomes Oca in-place
    float* Tfa_f = (float*)alloc(TN * 4);            // becomes Ofa in-place
    ush*   Tca_b = (ush*)alloc(TN * 2);
    ush*   Tfa_b = (ush*)alloc(TN * 2);
    float* Sbuf  = (float*)alloc(SN * 4);
    ush*   Attn2 = (ush*)alloc(SN * 2);

    auto gemm = [&](const void* A, const void* Bm, void* Cf, void* Cb,
                    const void* bias, int bias_mode, long long sBias, int zdBias,
                    int M, int Nn, int K, int lda, int ldb, int tA, int tB,
                    long long sA, int zdA, long long sB, int zdB, int Z, float alpha) {
        dim3 g((Nn + BN - 1) / BN, (M + BM - 1) / BM, Z);
        dim3 blk(256);
#define GEMM_ARGS (const ush*)A, (const ush*)Bm, (float*)Cf, (ush*)Cb, \
                  (const float*)bias, bias_mode, sBias, zdBias, \
                  M, Nn, K, lda, ldb, sA, zdA, sB, zdB, alpha
        if (tA == 0 && tB == 0)      wmma_gemm_bf16<0,0><<<g, blk, 0, stream>>>(GEMM_ARGS);
        else if (tA == 0 && tB == 1) wmma_gemm_bf16<0,1><<<g, blk, 0, stream>>>(GEMM_ARGS);
        else if (tA == 1 && tB == 1) wmma_gemm_bf16<1,1><<<g, blk, 0, stream>>>(GEMM_ARGS);
        else                         wmma_gemm_bf16<1,0><<<g, blk, 0, stream>>>(GEMM_ARGS);
#undef GEMM_ARGS
    };
    auto cast = [&](const float* in, ush* o, size_t n) {
        cast_f32_bf16_k<<<dim3((unsigned)((n + 255) / 256)), dim3(256), 0, stream>>>(in, o, n);
    };

    // -------- stage 0: cast activations + weights to bf16 --------
    cast(features, Xbf, XN);
    cast(sa_wq, Wsaq, (size_t)NV_*QK_*C_);
    cast(sa_wk, Wsak, (size_t)NV_*QK_*C_);
    cast(sa_wv, Wsav, (size_t)NV_*C_*C_);
    cast(ca_wq, Wcaq, (size_t)CD_*CD_);
    cast(ca_wk, Wcak, (size_t)CD_*CD_);
    cast(ca_wv, Wcav, (size_t)CD_*CD_);
    cast(fa_wq, Wfaq, (size_t)FD_*FD_);
    cast(fa_wk, Wfak, (size_t)FD_*FD_);
    cast(fa_wv, Wfav, (size_t)FD_*FD_);

    // -------- stage 1: per-view self-attention, Z = NV*B = 64, z = v*16+b --------
    const int ZSA = NV_ * B_;
    gemm(Xbf, Wsaq, nullptr, Qt, sa_bq, 2, QK_, B_,
         N_, QK_, C_, N_, C_, 1, 1,
         (long long)C_*N_, 1, (long long)QK_*C_, B_, ZSA, 1.0f);
    gemm(Xbf, Wsak, nullptr, Kt, sa_bk, 2, QK_, B_,
         N_, QK_, C_, N_, C_, 1, 1,
         (long long)C_*N_, 1, (long long)QK_*C_, B_, ZSA, 1.0f);
    gemm(Wsav, Xbf, nullptr, Vbf, sa_bv, 1, C_, B_,
         C_, N_, C_, C_, N_, 0, 0,
         (long long)C_*C_, B_, (long long)C_*N_, 1, ZSA, 1.0f);
    gemm(Qt, Kt, Epool, nullptr, nullptr, 0, 0, 1,
         N_, N_, QK_, QK_, QK_, 0, 1,
         (long long)N_*QK_, 1, (long long)N_*QK_, 1, ZSA, 1.0f);
    softmax_rows_k<<<dim3((unsigned)(ZSA * N_)), dim3(256), 0, stream>>>(Epool, AttnS, N_);
    gemm(Vbf, AttnS, Epool, nullptr, nullptr, 0, 0, 1,
         C_, N_, N_, N_, N_, 0, 1,
         (long long)C_*N_, 1, (long long)N_*N_, 1, ZSA, 1.0f);
    sa_epilogue_k<<<dim3((unsigned)((XN + 255) / 256)), dim3(256), 0, stream>>>(
        Epool, features, sa_gamma, (size_t)B_ * C_ * N_, XN);
    repack_sa_k<<<dim3((unsigned)((XN + 255) / 256)), dim3(256), 0, stream>>>(
        Epool, Tca_f, Tca_b, Tfa_f, Tfa_b);

    // -------- stage 2: feature_attention (R=768, D=3136), Z = 16 --------
    ush* Qfa = Xbf;
    ush* Kfa = Vbf;
    ush* Vfa = AttnS;
    gemm(Tfa_b, Wfaq, nullptr, Qfa, fa_bq, 2, 0, 1,
         C_, FD_, FD_, FD_, FD_, 0, 1, (long long)C_*FD_, 1, 0, 1, B_, 1.0f);
    gemm(Tfa_b, Wfak, nullptr, Kfa, fa_bk, 2, 0, 1,
         C_, FD_, FD_, FD_, FD_, 0, 1, (long long)C_*FD_, 1, 0, 1, B_, 1.0f);
    gemm(Tfa_b, Wfav, nullptr, Vfa, fa_bv, 2, 0, 1,
         C_, FD_, FD_, FD_, FD_, 0, 1, (long long)C_*FD_, 1, 0, 1, B_, 1.0f);
    gemm(Qfa, Kfa, Sbuf, nullptr, nullptr, 0, 0, 1,
         C_, C_, FD_, FD_, FD_, 0, 1,
         (long long)C_*FD_, 1, (long long)C_*FD_, 1, B_, 0.017857142857142856f); // 3136^-0.5
    softmax_rows_k<<<dim3((unsigned)(B_ * C_)), dim3(256), 0, stream>>>(Sbuf, Attn2, C_);
    gemm(Attn2, Vfa, Epool, nullptr, nullptr, 0, 0, 1,
         C_, FD_, C_, C_, FD_, 0, 0,
         (long long)C_*C_, 1, (long long)C_*FD_, 1, B_, 1.0f);
    scramble_residual_k<<<dim3((unsigned)((TN + 255) / 256)), dim3(256), 0, stream>>>(
        Epool, Tfa_f, C_, FD_, TN);   // Tfa_f is now Ofa

    // -------- stage 3: channel_attention (R=784, D=3072), Z = 16 --------
    ush* Qca = Qfa; ush* Kca = Kfa; ush* Vca = Vfa;
    gemm(Tca_b, Wcaq, nullptr, Qca, ca_bq, 2, 0, 1,
         N_, CD_, CD_, CD_, CD_, 0, 1, (long long)N_*CD_, 1, 0, 1, B_, 1.0f);
    gemm(Tca_b, Wcak, nullptr, Kca, ca_bk, 2, 0, 1,
         N_, CD_, CD_, CD_, CD_, 0, 1, (long long)N_*CD_, 1, 0, 1, B_, 1.0f);
    gemm(Tca_b, Wcav, nullptr, Vca, ca_bv, 2, 0, 1,
         N_, CD_, CD_, CD_, CD_, 0, 1, (long long)N_*CD_, 1, 0, 1, B_, 1.0f);
    gemm(Qca, Kca, Sbuf, nullptr, nullptr, 0, 0, 1,
         N_, N_, CD_, CD_, CD_, 0, 1,
         (long long)N_*CD_, 1, (long long)N_*CD_, 1, B_, 0.018042195912175807f); // 3072^-0.5
    softmax_rows_k<<<dim3((unsigned)(B_ * N_)), dim3(256), 0, stream>>>(Sbuf, Attn2, N_);
    gemm(Attn2, Vca, Epool, nullptr, nullptr, 0, 0, 1,
         N_, CD_, N_, N_, CD_, 0, 0,
         (long long)N_*N_, 1, (long long)N_*CD_, 1, B_, 1.0f);
    scramble_residual_k<<<dim3((unsigned)((TN + 255) / 256)), dim3(256), 0, stream>>>(
        Epool, Tca_f, N_, CD_, TN);   // Tca_f is now Oca

    // -------- stage 4: combine + mean over views --------
    final_combine_k<<<dim3((unsigned)(((size_t)B_ * C_ * N_ + 255) / 256)), dim3(256), 0, stream>>>(
        Tfa_f, Tca_f, out);
}
